// CrossAttentionTransformerBlock_41540923687503
// MI455X (gfx1250) — compile-verified, matrix operand order checked by reading, not run on hardware
//
#include <hip/hip_runtime.h>
#include <math.h>

// ---------------------------------------------------------------------------
// CDNA5 (gfx1250) bf16-WMMA implementation of the cross-attention transformer
// block.  All GEMMs and both attention einsums run through
// v_wmma_f32_16x16x32_bf16 (fp32 accumulate).  Softmax / LayerNorm / GELU in
// fp32.  Workspace requirement: ~180 MB.
// ---------------------------------------------------------------------------

typedef __attribute__((ext_vector_type(16))) __bf16 v16bf;
typedef __attribute__((ext_vector_type(8)))  float  v8f;
typedef __attribute__((ext_vector_type(4)))  float  f4;

static constexpr int Hn   = 768;
static constexpr int NHn  = 12;
static constexpr int SLn  = 1024;
static constexpr int SHn  = 2048;
static constexpr int Bn   = 2;
static constexpr int MLPn = 3072;
static constexpr int HDn  = 64;

__device__ __forceinline__ unsigned short f2bf(float f) {
    unsigned u = __float_as_uint(f);
    unsigned r = u + 0x7FFFu + ((u >> 16) & 1u);   // round-to-nearest-even
    return (unsigned short)(r >> 16);
}
__device__ __forceinline__ float bf2f(unsigned short h) {
    return __uint_as_float(((unsigned)h) << 16);
}

union Frag { v16bf v; f4 f[2]; };

// ---------------------------------------------------------------------------
// Weight convert + transpose: f32 [K][N] row-major -> bf16 [N][K] row-major.
// Makes the GEMM B-fragment a contiguous-K load (two ds_load_b128 per lane).
// ---------------------------------------------------------------------------
__global__ __launch_bounds__(256)
void cvt_w_transpose(const float* __restrict__ in, unsigned short* __restrict__ out,
                     int K, int N)
{
    const size_t idx = (size_t)blockIdx.x * 256 + threadIdx.x;
    if (idx >= (size_t)K * N) return;
    const int k = (int)(idx / N);
    const int n = (int)(idx % N);
    out[(size_t)n * K + k] = f2bf(in[idx]);
}

__global__ __launch_bounds__(256)
void cvt_f32_bf16(const float* __restrict__ in, unsigned short* __restrict__ out, size_t n)
{
    const size_t idx = (size_t)blockIdx.x * 256 + threadIdx.x;
    if (idx < n) out[idx] = f2bf(in[idx]);
}

// ---------------------------------------------------------------------------
// LayerNorm over H=768 (=3*256), fp32 in -> bf16 out. One block per row.
// ---------------------------------------------------------------------------
__global__ __launch_bounds__(256)
void layernorm_bf16(const float* __restrict__ x, const float* __restrict__ g,
                    const float* __restrict__ bta, unsigned short* __restrict__ out)
{
    __shared__ float red[256];
    const int row = blockIdx.x, tid = threadIdx.x;
    const float* xr = x + (size_t)row * Hn;
    float s = 0.f, s2 = 0.f;
    #pragma unroll
    for (int i = 0; i < 3; ++i) { float v = xr[tid + i * 256]; s += v; s2 += v * v; }
    red[tid] = s; __syncthreads();
    for (int o = 128; o > 0; o >>= 1) { if (tid < o) red[tid] += red[tid + o]; __syncthreads(); }
    const float mean = red[0] * (1.0f / Hn);
    __syncthreads();
    red[tid] = s2; __syncthreads();
    for (int o = 128; o > 0; o >>= 1) { if (tid < o) red[tid] += red[tid + o]; __syncthreads(); }
    const float var = red[0] * (1.0f / Hn) - mean * mean;
    const float inv = rsqrtf(var + 1e-5f);
    #pragma unroll
    for (int i = 0; i < 3; ++i) {
        const int c = tid + i * 256;
        out[(size_t)row * Hn + c] = f2bf((xr[c] - mean) * inv * g[c] + bta[c]);
    }
}

// ---------------------------------------------------------------------------
// Tiled bf16 GEMM: C[M][Nout] = A[M][K] * Wt[n0+n][K]^T  (Wt is [Nfull][K]).
// 128x128x32 tile, 256 threads = 8 wave32 in a 4(M) x 2(N) grid; each wave
// owns 2x4 16x16 fragments -> 8 v_wmma_f32_16x16x32_bf16 per K-step.
// Epilogue: optional bias, exact GELU, fp32 residual; fp32 or bf16 output.
// ---------------------------------------------------------------------------
template<bool OUT_BF16, bool HAS_BIAS, bool DO_GELU, bool HAS_RES>
__global__ __launch_bounds__(256)
void gemm_bf16_wmma(const unsigned short* __restrict__ A,
                    const unsigned short* __restrict__ Wt,
                    const float* __restrict__ bias,
                    const float* __restrict__ res,
                    float* __restrict__ outF,
                    unsigned short* __restrict__ outB,
                    int M, int K, int n0, int Nout)
{
    constexpr int BM = 128, BN = 128, BK = 32, LDT = 48;   // 96B row: 16B aligned
    __shared__ unsigned short As[BM * LDT];
    __shared__ unsigned short Bs[BN * LDT];

    const int tid  = threadIdx.x;
    const int row0 = blockIdx.y * BM;
    const int col0 = blockIdx.x * BN;
    const int wave = tid >> 5, lane = tid & 31;
    const int wm   = wave & 3;          // 4 waves along M (32 rows each)
    const int wn   = wave >> 2;         // 2 waves along N (64 cols each)
    const int lhi  = lane >> 4;         // K-half select per ISA A/B layout
    const int llo  = lane & 15;

    const v8f vzero = {0.f,0.f,0.f,0.f,0.f,0.f,0.f,0.f};
    v8f acc[2][4];
    #pragma unroll
    for (int i = 0; i < 2; ++i)
        #pragma unroll
        for (int j = 0; j < 4; ++j) acc[i][j] = vzero;

    for (int kt = 0; kt < K; kt += BK) {
        #pragma unroll
        for (int i = 0; i < 2; ++i) {                 // 512 chunks of 8 bf16
            const int ch = tid + i * 256;
            const int r  = ch >> 2;                   // 0..127
            const int kc = (ch & 3) << 3;             // 0/8/16/24
            *(f4*)&As[r * LDT + kc] = *(const f4*)&A [(size_t)(row0 + r)      * K + kt + kc];
            *(f4*)&Bs[r * LDT + kc] = *(const f4*)&Wt[(size_t)(n0 + col0 + r) * K + kt + kc];
        }
        if (kt + BK < K) {                            // global_prefetch_b8 of next tiles
            __builtin_prefetch(&A [(size_t)(row0 + (tid >> 1))      * K + kt + BK], 0, 0);
            __builtin_prefetch(&Wt[(size_t)(n0 + col0 + (tid >> 1)) * K + kt + BK], 0, 0);
        }
        __syncthreads();

        Frag a[2], b[4];
        #pragma unroll
        for (int mi = 0; mi < 2; ++mi) {
            const unsigned short* p = &As[(wm * 32 + mi * 16 + llo) * LDT + lhi * 8];
            a[mi].f[0] = *(const f4*)p;
            a[mi].f[1] = *(const f4*)(p + 16);
        }
        #pragma unroll
        for (int ni = 0; ni < 4; ++ni) {
            const unsigned short* p = &Bs[(wn * 64 + ni * 16 + llo) * LDT + lhi * 8];
            b[ni].f[0] = *(const f4*)p;
            b[ni].f[1] = *(const f4*)(p + 16);
        }
        #pragma unroll
        for (int mi = 0; mi < 2; ++mi)
            #pragma unroll
            for (int ni = 0; ni < 4; ++ni)
                acc[mi][ni] = __builtin_amdgcn_wmma_f32_16x16x32_bf16(
                    false, a[mi].v, false, b[ni].v, (short)0, acc[mi][ni], false, false);
        __syncthreads();
    }

    #pragma unroll
    for (int mi = 0; mi < 2; ++mi) {
        #pragma unroll
        for (int ni = 0; ni < 4; ++ni) {
            #pragma unroll
            for (int r = 0; r < 8; ++r) {
                const int row = row0 + wm * 32 + mi * 16 + lhi * 8 + r;
                const int col = col0 + wn * 64 + ni * 16 + llo;
                float v = acc[mi][ni][r];
                if (HAS_BIAS) v += bias[col];
                if (DO_GELU)  v = 0.5f * v * (1.0f + erff(v * 0.70710678118654752f));
                const size_t idx = (size_t)row * Nout + col;
                if (HAS_RES)  v += res[idx];
                if (OUT_BF16) outB[idx] = f2bf(v);
                else          outF[idx] = v;
            }
        }
    }
}

// ---------------------------------------------------------------------------
// Flash-style fused attention (online softmax) for one (batch, head, q-block).
// 64 q-rows x 64 kv tiles; S in LDS fp32; P re-quantized to bf16 and fed back
// through WMMA for P*V (V tile stored transposed so B-fragment is contiguous-K).
// QRES adds the per-head q residual (cross-attention: o = q + att @ v).
// ---------------------------------------------------------------------------
template<bool QRES>
__global__ __launch_bounds__(256)
void flash_attn_wmma(const unsigned short* __restrict__ Qb, int qStride, int qOff0,
                     const unsigned short* __restrict__ Kb, int kStride, int kOff0,
                     const unsigned short* __restrict__ Vb, int vStride, int vOff0,
                     unsigned short* __restrict__ Ob,
                     int Sq, int Sk, float scale)
{
    constexpr int BQ = 64, BKV = 64, LD = 80, SLD = 66;   // 160B rows: 16B aligned
    __shared__ unsigned short Qs [BQ  * LD];
    __shared__ unsigned short Ks [BKV * LD];
    __shared__ unsigned short Vts[HDn * LD];              // [d][kv] (transposed)
    __shared__ float          Ss [BQ * SLD];
    __shared__ unsigned short Ps [BQ * LD];
    __shared__ float row_m[BQ], row_l[BQ], row_a[BQ];

    const int tid = threadIdx.x;
    const int q0  = blockIdx.x * BQ;
    const int h   = blockIdx.y;
    const int bb  = blockIdx.z;
    const int wave = tid >> 5, lane = tid & 31;
    const int wm = wave & 1, wn = wave >> 1;              // 2(M=32) x 4(N=16)
    const int lhi = lane >> 4, llo = lane & 15;

    const int qOff = qOff0 + h * HDn;
    const int kOff = kOff0 + h * HDn;
    const int vOff = vOff0 + h * HDn;
    const size_t qRow0 = (size_t)bb * Sq + q0;
    const size_t kRow0 = (size_t)bb * Sk;

    #pragma unroll
    for (int i = 0; i < 2; ++i) {                         // load Q tile 64x64
        const int ch = tid + i * 256;
        const int qr = ch >> 3;
        const int dc = (ch & 7) << 3;
        *(f4*)&Qs[qr * LD + dc] = *(const f4*)&Qb[(qRow0 + qr) * qStride + qOff + dc];
    }
    if (tid < BQ) { row_m[tid] = -3.0e38f; row_l[tid] = 0.f; }

    const v8f vzero = {0.f,0.f,0.f,0.f,0.f,0.f,0.f,0.f};
    v8f o[2] = {vzero, vzero};
    __syncthreads();

    for (int nk = 0; nk < Sk; nk += BKV) {
        #pragma unroll
        for (int i = 0; i < 2; ++i) {                     // load K, V(transposed)
            const int ch = tid + i * 256;
            const int kv = ch >> 3;
            const int dc = (ch & 7) << 3;
            *(f4*)&Ks[kv * LD + dc] = *(const f4*)&Kb[(kRow0 + nk + kv) * kStride + kOff + dc];
            f4 vv = *(const f4*)&Vb[(kRow0 + nk + kv) * vStride + vOff + dc];
            #pragma unroll
            for (int j = 0; j < 4; ++j) {
                const unsigned u = __float_as_uint(vv[j]);
                Vts[(dc + 2 * j    ) * LD + kv] = (unsigned short)(u & 0xFFFFu);
                Vts[(dc + 2 * j + 1) * LD + kv] = (unsigned short)(u >> 16);
            }
        }
        __syncthreads();

        // ---- S = Q * K^T  (K-dim = 64 -> two 16x16x32 WMMAs per fragment)
        v8f s[2] = {vzero, vzero};
        #pragma unroll
        for (int kk = 0; kk < HDn; kk += 32) {
            Frag bk;
            { const unsigned short* p = &Ks[(wn * 16 + llo) * LD + kk + lhi * 8];
              bk.f[0] = *(const f4*)p; bk.f[1] = *(const f4*)(p + 16); }
            #pragma unroll
            for (int mi = 0; mi < 2; ++mi) {
                Frag aq;
                const unsigned short* p = &Qs[(wm * 32 + mi * 16 + llo) * LD + kk + lhi * 8];
                aq.f[0] = *(const f4*)p; aq.f[1] = *(const f4*)(p + 16);
                s[mi] = __builtin_amdgcn_wmma_f32_16x16x32_bf16(
                    false, aq.v, false, bk.v, (short)0, s[mi], false, false);
            }
        }
        #pragma unroll
        for (int mi = 0; mi < 2; ++mi)
            #pragma unroll
            for (int r = 0; r < 8; ++r)
                Ss[(wm * 32 + mi * 16 + lhi * 8 + r) * SLD + wn * 16 + llo] = s[mi][r] * scale;
        __syncthreads();

        // ---- online softmax (one thread per q row)
        if (tid < BQ) {
            const float mo = row_m[tid];
            float mx = mo;
            for (int c = 0; c < BKV; ++c) mx = fmaxf(mx, Ss[tid * SLD + c]);
            const float alpha = __expf(mo - mx);
            float lsum = row_l[tid] * alpha;
            for (int c = 0; c < BKV; ++c) {
                const float e = __expf(Ss[tid * SLD + c] - mx);
                Ps[tid * LD + c] = f2bf(e);
                lsum += e;
            }
            row_m[tid] = mx; row_l[tid] = lsum; row_a[tid] = alpha;
        }
        __syncthreads();

        // ---- rescale O, then O += P * V
        #pragma unroll
        for (int mi = 0; mi < 2; ++mi)
            #pragma unroll
            for (int r = 0; r < 8; ++r)
                o[mi][r] = o[mi][r] * row_a[wm * 32 + mi * 16 + lhi * 8 + r];
        #pragma unroll
        for (int kk = 0; kk < BKV; kk += 32) {
            Frag bv;
            { const unsigned short* p = &Vts[(wn * 16 + llo) * LD + kk + lhi * 8];
              bv.f[0] = *(const f4*)p; bv.f[1] = *(const f4*)(p + 16); }
            #pragma unroll
            for (int mi = 0; mi < 2; ++mi) {
                Frag ap;
                const unsigned short* p = &Ps[(wm * 32 + mi * 16 + llo) * LD + kk + lhi * 8];
                ap.f[0] = *(const f4*)p; ap.f[1] = *(const f4*)(p + 16);
                o[mi] = __builtin_amdgcn_wmma_f32_16x16x32_bf16(
                    false, ap.v, false, bv.v, (short)0, o[mi], false, false);
            }
        }
        __syncthreads();
    }

    // ---- finalize: O /= l, optional +q residual, write [b,s,H] bf16
    #pragma unroll
    for (int mi = 0; mi < 2; ++mi) {
        #pragma unroll
        for (int r = 0; r < 8; ++r) {
            const int row = wm * 32 + mi * 16 + lhi * 8 + r;
            const int d   = wn * 16 + llo;
            float v = o[mi][r] / row_l[row];
            if (QRES) v += bf2f(Qs[row * LD + d]);
            Ob[(qRow0 + row) * (size_t)Hn + h * HDn + d] = f2bf(v);
        }
    }
}

// ---------------------------------------------------------------------------
// Host-side orchestration
// ---------------------------------------------------------------------------
extern "C" void kernel_launch(void* const* d_in, const int* in_sizes, int n_in,
                              void* d_out, int out_size, void* d_ws, size_t ws_size,
                              hipStream_t stream)
{
    (void)in_sizes; (void)n_in; (void)out_size; (void)ws_size;

    const float* lx     = (const float*)d_in[0];
    const float* hx     = (const float*)d_in[1];
    const float* Wqkv_l = (const float*)d_in[2];
    const float* Wo_l   = (const float*)d_in[3];
    const float* bo_l   = (const float*)d_in[4];
    const float* Wqkv_h = (const float*)d_in[5];
    const float* Wo_h   = (const float*)d_in[6];
    const float* bo_h   = (const float*)d_in[7];
    const float* Wqkv_c = (const float*)d_in[8];
    const float* Wo_c   = (const float*)d_in[9];
    const float* bo_c   = (const float*)d_in[10];
    const float* g1     = (const float*)d_in[11];
    const float* b1     = (const float*)d_in[12];
    const float* g2     = (const float*)d_in[13];
    const float* b2     = (const float*)d_in[14];
    const float* W1     = (const float*)d_in[15];
    const float* bm1    = (const float*)d_in[16];
    const float* W2     = (const float*)d_in[17];
    const float* bm2    = (const float*)d_in[18];

    const int ML = Bn * SLn;   // 2048
    const int MH = Bn * SHn;   // 4096
    const size_t A_l = (size_t)ML * Hn, A_h = (size_t)MH * Hn;
    const float SCALE = 0.125f;  // HD^-0.5

    // -------- workspace carve-up (~180 MB total) --------
    char* base = (char*)d_ws;
    size_t off = 0;
    auto alloc = [&](size_t bytes) -> void* {
        off = (off + 255) & ~(size_t)255;
        void* p = base + off;
        off += bytes;
        return p;
    };
    auto aU = [&](size_t e) { return (unsigned short*)alloc(e * 2); };
    auto aF = [&](size_t e) { return (float*)alloc(e * 4); };

    unsigned short* wt_qkv_l = aU((size_t)3 * Hn * Hn);   // [2304][768] bf16
    unsigned short* wt_qkv_h = aU((size_t)3 * Hn * Hn);
    unsigned short* wt_qkv_c = aU((size_t)3 * Hn * Hn);
    unsigned short* wt_o_l   = aU((size_t)Hn * Hn);       // [768][768]
    unsigned short* wt_o_h   = aU((size_t)Hn * Hn);
    unsigned short* wt_o_c   = aU((size_t)Hn * Hn);
    unsigned short* wt_1     = aU((size_t)MLPn * Hn);     // [3072][768]
    unsigned short* wt_2     = aU((size_t)Hn * MLPn);     // [768][3072]

    unsigned short* ln_l  = aU(A_l);
    unsigned short* ln_h  = aU(A_h);
    unsigned short* qkvl  = aU((size_t)ML * 3 * Hn);
    unsigned short* qkvh  = aU((size_t)MH * 3 * Hn);
    unsigned short* oatl  = aU(A_l);
    unsigned short* oath  = aU(A_h);
    float*          lxa_f = aF(A_l);
    float*          hxa_f = aF(A_h);
    unsigned short* lxa_b = aU(A_l);
    unsigned short* hxa_b = aU(A_h);
    unsigned short* qc    = aU(A_l);
    unsigned short* kvc   = aU((size_t)MH * 2 * Hn);
    unsigned short* oc    = aU(A_l);
    float*          lxc_f = aF(A_l);
    unsigned short* ln2l  = aU(A_l);
    unsigned short* ln2h  = aU(A_h);
    unsigned short* h1l   = aU((size_t)ML * MLPn);
    unsigned short* h1h   = aU((size_t)MH * MLPn);

    float* out_lx = (float*)d_out;          // [2,1024,768]
    float* out_hx = out_lx + A_l;           // [2,2048,768]

    // -------- 1) convert + transpose all weights to bf16 [N][K] --------
    auto cvtW = [&](const float* w, unsigned short* wt, int K, int N) {
        const size_t tot = (size_t)K * N;
        cvt_w_transpose<<<(unsigned)((tot + 255) / 256), 256, 0, stream>>>(w, wt, K, N);
    };
    cvtW(Wqkv_l, wt_qkv_l, Hn, 3 * Hn);
    cvtW(Wqkv_h, wt_qkv_h, Hn, 3 * Hn);
    cvtW(Wqkv_c, wt_qkv_c, Hn, 3 * Hn);
    cvtW(Wo_l,   wt_o_l,   Hn, Hn);
    cvtW(Wo_h,   wt_o_h,   Hn, Hn);
    cvtW(Wo_c,   wt_o_c,   Hn, Hn);
    cvtW(W1,     wt_1,     Hn, MLPn);
    cvtW(W2,     wt_2,     MLPn, Hn);

    // -------- 2) LayerNorm-1 (shared g1/b1) --------
    layernorm_bf16<<<ML, 256, 0, stream>>>(lx, g1, b1, ln_l);
    layernorm_bf16<<<MH, 256, 0, stream>>>(hx, g1, b1, ln_h);

    // -------- 3) QKV projections (no bias, bf16 out) --------
    gemm_bf16_wmma<true, false, false, false><<<dim3(3 * Hn / 128, ML / 128), 256, 0, stream>>>(
        ln_l, wt_qkv_l, nullptr, nullptr, nullptr, qkvl, ML, Hn, 0, 3 * Hn);
    gemm_bf16_wmma<true, false, false, false><<<dim3(3 * Hn / 128, MH / 128), 256, 0, stream>>>(
        ln_h, wt_qkv_h, nullptr, nullptr, nullptr, qkvh, MH, Hn, 0, 3 * Hn);

    // -------- 4) self-attention (fused flash) --------
    flash_attn_wmma<false><<<dim3(SLn / 64, NHn, Bn), 256, 0, stream>>>(
        qkvl, 3 * Hn, 0, qkvl, 3 * Hn, Hn, qkvl, 3 * Hn, 2 * Hn, oatl, SLn, SLn, SCALE);
    flash_attn_wmma<false><<<dim3(SHn / 64, NHn, Bn), 256, 0, stream>>>(
        qkvh, 3 * Hn, 0, qkvh, 3 * Hn, Hn, qkvh, 3 * Hn, 2 * Hn, oath, SHn, SHn, SCALE);

    // -------- 5) output projections (+bias, fp32) + bf16 copies --------
    gemm_bf16_wmma<false, true, false, false><<<dim3(Hn / 128, ML / 128), 256, 0, stream>>>(
        oatl, wt_o_l, bo_l, nullptr, lxa_f, nullptr, ML, Hn, 0, Hn);
    gemm_bf16_wmma<false, true, false, false><<<dim3(Hn / 128, MH / 128), 256, 0, stream>>>(
        oath, wt_o_h, bo_h, nullptr, hxa_f, nullptr, MH, Hn, 0, Hn);
    cvt_f32_bf16<<<(unsigned)((A_l + 255) / 256), 256, 0, stream>>>(lxa_f, lxa_b, A_l);
    cvt_f32_bf16<<<(unsigned)((A_h + 255) / 256), 256, 0, stream>>>(hxa_f, hxa_b, A_h);

    // -------- 6) cross QKV: q from lxa (cols 0..768), kv from hxa (768..2304) --------
    gemm_bf16_wmma<true, false, false, false><<<dim3(Hn / 128, ML / 128), 256, 0, stream>>>(
        lxa_b, wt_qkv_c, nullptr, nullptr, nullptr, qc, ML, Hn, 0, Hn);
    gemm_bf16_wmma<true, false, false, false><<<dim3(2 * Hn / 128, MH / 128), 256, 0, stream>>>(
        hxa_b, wt_qkv_c, nullptr, nullptr, nullptr, kvc, MH, Hn, Hn, 2 * Hn);

    // -------- 7) cross-attention with q residual (o = q + att @ v) --------
    flash_attn_wmma<true><<<dim3(SLn / 64, NHn, Bn), 256, 0, stream>>>(
        qc, Hn, 0, kvc, 2 * Hn, 0, kvc, 2 * Hn, Hn, oc, SLn, SHn, SCALE);

    // -------- 8) cross out-projection -> lxc (fp32) --------
    gemm_bf16_wmma<false, true, false, false><<<dim3(Hn / 128, ML / 128), 256, 0, stream>>>(
        oc, wt_o_c, bo_c, nullptr, lxc_f, nullptr, ML, Hn, 0, Hn);

    // -------- 9) LayerNorm-2 (shared g2/b2) --------
    layernorm_bf16<<<ML, 256, 0, stream>>>(lxc_f, g2, b2, ln2l);
    layernorm_bf16<<<MH, 256, 0, stream>>>(hxa_f, g2, b2, ln2h);

    // -------- 10) MLP: W1 (+bm1, GELU fused) then W2 (+bm2, +residual) -> d_out --------
    gemm_bf16_wmma<true, true, true, false><<<dim3(MLPn / 128, ML / 128), 256, 0, stream>>>(
        ln2l, wt_1, bm1, nullptr, nullptr, h1l, ML, Hn, 0, MLPn);
    gemm_bf16_wmma<true, true, true, false><<<dim3(MLPn / 128, MH / 128), 256, 0, stream>>>(
        ln2h, wt_1, bm1, nullptr, nullptr, h1h, MH, Hn, 0, MLPn);
    gemm_bf16_wmma<false, true, false, true><<<dim3(Hn / 128, ML / 128), 256, 0, stream>>>(
        h1l, wt_2, bm2, lxc_f, out_lx, nullptr, ML, MLPn, 0, Hn);
    gemm_bf16_wmma<false, true, false, true><<<dim3(Hn / 128, MH / 128), 256, 0, stream>>>(
        h1h, wt_2, bm2, hxa_f, out_hx, nullptr, MH, MLPn, 0, Hn);
}
